// Distance_layer_40819369181533
// MI455X (gfx1250) — matrix operand order repaired and checked by reading, not compiled, not run on hardware
//
#include <hip/hip_runtime.h>

typedef __attribute__((ext_vector_type(16))) __bf16 v16bf;
typedef __attribute__((ext_vector_type(8)))  __bf16 v8bf;
typedef __attribute__((ext_vector_type(8)))  float  v8f;

#define TILE_M 128
#define TILE_N 128
#define TILE_K 32
#define LDSS   40   // 32 + 8 pad bf16 elems -> 80B row stride, 16B aligned, conflict-free

__device__ __forceinline__ v16bf concat8(v8bf lo, v8bf hi) {
  v16bf r;
#pragma unroll
  for (int i = 0; i < 8; ++i) { r[i] = lo[i]; r[i + 8] = hi[i]; }
  return r;
}

__global__ __launch_bounds__(256)
void l2dist_wmma_kernel(const float* __restrict__ x,
                        const float* __restrict__ w,
                        float* __restrict__ out,
                        int N, int P, int F)
{
  // Double-buffered bf16 hi/lo tiles: 4 arrays * 2 bufs * 10240B = ~80KB of 320KB WGP LDS.
  __shared__ __attribute__((aligned(16))) __bf16 sXh[2][TILE_M * LDSS];
  __shared__ __attribute__((aligned(16))) __bf16 sXl[2][TILE_M * LDSS];
  __shared__ __attribute__((aligned(16))) __bf16 sWh[2][TILE_N * LDSS];
  __shared__ __attribute__((aligned(16))) __bf16 sWl[2][TILE_N * LDSS];
  __shared__ float sXsq[TILE_M];
  __shared__ float sWsq[TILE_N];

  const int tid    = threadIdx.x;
  const int lane   = tid & 31;
  const int laneLo = lane & 15;
  const int laneHi = lane >> 4;          // half-wave select
  const int wave   = tid >> 5;
  const int waveM  = wave >> 1;          // 0..3 -> 32 output rows each
  const int waveN  = wave & 1;           // 0..1 -> 64 output cols each

  const int rowBase = blockIdx.x * TILE_M;
  const int colBase = blockIdx.y * TILE_N;

  // Staging: 256 threads cover 128 rows x 32 K (16 f32 per thread per matrix)
  const int tr = tid >> 1;
  const int tc = (tid & 1) * 16;

  const float* xg = x + (size_t)(rowBase + tr) * F + tc;
  const float* wg = w + (size_t)(colBase + tr) * F + tc;

  if (tid < TILE_M) sXsq[tid] = 0.f; else sWsq[tid - TILE_M] = 0.f;

  v8f acc[2][4];
#pragma unroll
  for (int mt = 0; mt < 2; ++mt)
#pragma unroll
    for (int nt = 0; nt < 4; ++nt)
      acc[mt][nt] = (v8f){};

  // Pipeline prologue: global loads for K-step 0.
  float4 xv[4], wv[4];
#pragma unroll
  for (int i = 0; i < 4; ++i) {
    xv[i] = *(const float4*)(xg + 4 * i);
    wv[i] = *(const float4*)(wg + 4 * i);
  }

  __syncthreads();   // norm-array init visible before first ds_add_f32

  const int nk = F / TILE_K;   // 16
  for (int kk = 0; kk < nk; ++kk) {
    const int buf = kk & 1;

    // ---- stage: split f32 -> bf16 hi + lo into LDS; fused row norms ----
    float xs = 0.f, ws = 0.f;
    v8bf xh[2], xl[2], wh[2], wl[2];
#pragma unroll
    for (int i = 0; i < 16; ++i) {
      float xf = ((const float*)xv)[i];
      float wf = ((const float*)wv)[i];
      xs += xf * xf;
      ws += wf * wf;
      __bf16 hx = (__bf16)xf;
      xh[i >> 3][i & 7] = hx;
      xl[i >> 3][i & 7] = (__bf16)(xf - (float)hx);
      __bf16 hw = (__bf16)wf;
      wh[i >> 3][i & 7] = hw;
      wl[i >> 3][i & 7] = (__bf16)(wf - (float)hw);
    }
    const int sb = tr * LDSS + tc;
    *(v8bf*)(&sXh[buf][sb])     = xh[0];
    *(v8bf*)(&sXh[buf][sb + 8]) = xh[1];
    *(v8bf*)(&sXl[buf][sb])     = xl[0];
    *(v8bf*)(&sXl[buf][sb + 8]) = xl[1];
    *(v8bf*)(&sWh[buf][sb])     = wh[0];
    *(v8bf*)(&sWh[buf][sb + 8]) = wh[1];
    *(v8bf*)(&sWl[buf][sb])     = wl[0];
    *(v8bf*)(&sWl[buf][sb + 8]) = wl[1];
    atomicAdd(&sXsq[tr], xs);   // ds_add_f32
    atomicAdd(&sWsq[tr], ws);

    __syncthreads();   // single barrier per iteration (double-buffered tiles)

    // ---- pipeline: issue next K-slab's global loads; latency hides under WMMAs ----
    if (kk + 1 < nk) {
      const int kn = (kk + 1) * TILE_K;
#pragma unroll
      for (int i = 0; i < 4; ++i) {
        xv[i] = *(const float4*)(xg + kn + 4 * i);
        wv[i] = *(const float4*)(wg + kn + 4 * i);
      }
      if (kk + 2 < nk) {
        __builtin_prefetch(xg + kn + TILE_K, 0, 0);   // global_prefetch_b8
        __builtin_prefetch(wg + kn + TILE_K, 0, 0);
      }
    }

    // ---- preload ALL fragments (one LDS clause, one wait), then 24 WMMAs ----
    // A (16x32 bf16): lane row=laneLo, K = {laneHi*8..+7, laneHi*8+16..+23}
    v16bf ah[2], al[2], bh[4], bl[4];
#pragma unroll
    for (int mt = 0; mt < 2; ++mt) {
      int r = (waveM * 32 + mt * 16 + laneLo) * LDSS + laneHi * 8;
      ah[mt] = concat8(*(const v8bf*)(&sXh[buf][r]), *(const v8bf*)(&sXh[buf][r + 16]));
      al[mt] = concat8(*(const v8bf*)(&sXl[buf][r]), *(const v8bf*)(&sXl[buf][r + 16]));
    }
    // B (32x16 bf16): lane col=laneLo, K = {laneHi*8..+7, laneHi*8+16..+23}
#pragma unroll
    for (int nt = 0; nt < 4; ++nt) {
      int c = (waveN * 64 + nt * 16 + laneLo) * LDSS + laneHi * 8;
      bh[nt] = concat8(*(const v8bf*)(&sWh[buf][c]), *(const v8bf*)(&sWh[buf][c + 16]));
      bl[nt] = concat8(*(const v8bf*)(&sWl[buf][c]), *(const v8bf*)(&sWl[buf][c + 16]));
    }
#pragma unroll
    for (int nt = 0; nt < 4; ++nt) {
#pragma unroll
      for (int mt = 0; mt < 2; ++mt) {
        // split-bf16 f32 emulation: hh + lh + hl (drop ll, second order)
        acc[mt][nt] = __builtin_amdgcn_wmma_f32_16x16x32_bf16(
            false, ah[mt], false, bh[nt], (short)0, acc[mt][nt], false, false);
        acc[mt][nt] = __builtin_amdgcn_wmma_f32_16x16x32_bf16(
            false, al[mt], false, bh[nt], (short)0, acc[mt][nt], false, false);
        acc[mt][nt] = __builtin_amdgcn_wmma_f32_16x16x32_bf16(
            false, ah[mt], false, bl[nt], (short)0, acc[mt][nt], false, false);
      }
    }
  }

  __syncthreads();   // sXsq / sWsq complete

  // Epilogue: out = |x|^2 + |w|^2 - 2*cross, streamed with non-temporal stores
  // (output is write-once; keep L2 for x-row reuse across column-blocks).
  // C/D layout: VGPR j, lanes 0-15 -> M=j, N=lane; lanes 16-31 -> M=j+8, N=lane-16.
#pragma unroll
  for (int nt = 0; nt < 4; ++nt) {
    int scol = waveN * 64 + nt * 16 + laneLo;
    float wsq = sWsq[scol];
    size_t col = (size_t)(colBase + scol);
#pragma unroll
    for (int mt = 0; mt < 2; ++mt) {
#pragma unroll
      for (int j = 0; j < 8; ++j) {
        int srow = waveM * 32 + mt * 16 + laneHi * 8 + j;
        float v = sXsq[srow] + wsq - 2.0f * acc[mt][nt][j];
        __builtin_nontemporal_store(v, out + (size_t)(rowBase + srow) * P + col);
      }
    }
  }
}

extern "C" void kernel_launch(void* const* d_in, const int* in_sizes, int n_in,
                              void* d_out, int out_size, void* d_ws, size_t ws_size,
                              hipStream_t stream) {
  const float* x = (const float*)d_in[0];   // [N, F] f32
  const float* w = (const float*)d_in[1];   // [P, F] f32
  float* out = (float*)d_out;               // [N, P] f32

  const int F = 512;
  const int N = in_sizes[0] / F;            // 65536
  const int P = in_sizes[1] / F;            // 512

  dim3 grid(N / TILE_M, P / TILE_N);        // (512, 4); grid.x fastest -> x rows reused in L2
  l2dist_wmma_kernel<<<grid, 256, 0, stream>>>(x, w, out, N, P, F);
}
